// Deformation_5068061409687
// MI455X (gfx1250) — matrix operand order repaired and checked by reading, not compiled, not run on hardware
//
#include <hip/hip_runtime.h>

typedef __attribute__((ext_vector_type(16))) _Float16 v16h;
typedef __attribute__((ext_vector_type(8)))  _Float16 v8h;
typedef __attribute__((ext_vector_type(8)))  float    v8f;
typedef __attribute__((ext_vector_type(4)))  unsigned int u32x4;
typedef __attribute__((ext_vector_type(8)))  unsigned int u32x8;

#define PLANE_R 128
#define PLANE_F 32

// Weight blob in d_ws (halfs): w_enc frags [0,8192), w1 [8192,73728), w2 [73728,86016)
#define WENC_OFF 0
#define W1_OFF   8192
#define W2_OFF   73728
#define WS_HALFS 86016
#define WS_BYTES (WS_HALFS * 2)          // 172032, divisible by 8
#define WS_E8    (WS_BYTES / 8)          // 21504 8-byte elements for TDM

// LDS layout (bytes), all 32B aligned. Total 325632 <= 320KB WGP LDS.
#define SX_STRIDE 36                      // halfs per X row (18-bank skew, conflict-free)
#define SH_STRIDE 264                     // halfs per H row (132-bank skew, conflict-free)
#define SW_LDS_OFF 0
#define SX_LDS_OFF WS_BYTES                                 // 172032
#define SH_LDS_OFF (SX_LDS_OFF + 256 * SX_STRIDE * 2)       // 190464
#define SMEM_BYTES (SH_LDS_OFF + 8 * 32 * SH_STRIDE * 2)    // 325632

__device__ __forceinline__ float clampf(float v, float lo, float hi) {
    return fminf(fmaxf(v, lo), hi);
}

// ---------------- weight convert + fragment swizzle ----------------
// Fragment = 512 halfs = one 32x16 (KxN) f16 B tile.
// Logical per-lane data: lane l owns 16 halfs j=0..15 with
//   n = nt*16 + (l&15),  k = kt*32 + ((l>>4)<<4) + j
// stored as 16B units u = 2*l + (j>>3), placed at XOR-swizzled position
//   pos = u ^ (u>>4)   (involution; makes ds_load_b128 2-way max = optimal)
__device__ __forceinline__ void swz_one(const float* src, _Float16* dst, int li,
                                        int NT, int Nw) {
    int frag = li >> 9;
    int r    = li & 511;                 // destination half within fragment
    int pos  = r >> 3;                   // 16B unit position 0..63
    int j7   = r & 7;
    int u    = pos ^ (pos >> 4);         // logical unit (involution)
    int lane = u >> 1;
    int j    = ((u & 1) << 3) + j7;
    int nt   = frag % NT;
    int kt   = frag / NT;
    int n    = nt * 16 + (lane & 15);
    int k    = kt * 32 + ((lane >> 4) << 4) + j;
    dst[li] = (_Float16)src[k * Nw + n];
}

__global__ void wconvert_kernel(const float* __restrict__ w_enc,
                                const float* __restrict__ w1,
                                const float* __restrict__ w2,
                                _Float16* __restrict__ ws) {
    int idx = blockIdx.x * blockDim.x + threadIdx.x;
    if (idx < W1_OFF) {
        swz_one(w_enc, ws + WENC_OFF, idx - WENC_OFF, 16, 256);   // 32x256
    } else if (idx < W2_OFF) {
        swz_one(w1, ws + W1_OFF, idx - W1_OFF, 16, 256);          // 256x256
    } else if (idx < WS_HALFS) {
        swz_one(w2, ws + W2_OFF, idx - W2_OFF, 3, 48);            // 256x48
    }
}

__global__ void passthrough_kernel(const float* __restrict__ pts,
                                   const float* __restrict__ rot,
                                   const float* __restrict__ h_emb,
                                   float* __restrict__ out, int N) {
    int i = blockIdx.x * blockDim.x + threadIdx.x;
    if (i >= N) return;
    out[3 * i + 0] = pts[3 * i + 0];
    out[3 * i + 1] = pts[3 * i + 1];
    out[3 * i + 2] = pts[3 * i + 2];
    float* orot = out + (size_t)3 * N;
    orot[4 * i + 0] = rot[4 * i + 0];
    orot[4 * i + 1] = rot[4 * i + 1];
    orot[4 * i + 2] = rot[4 * i + 2];
    orot[4 * i + 3] = rot[4 * i + 3];
    float* oh = out + (size_t)7 * N;
    oh[i] = h_emb[2 * i];
}

// A fragment (16x32 f16 WMMA A operand in contiguous k order, same as the B frags) from an LDS row pointer.
// lane l: M = l&15; halfs row[kb..kb+15], kb = (l>>4)*16
__device__ __forceinline__ v16h load_afrag(const _Float16* rowp, int kb) {
    v8h lo = *(const v8h*)(rowp + kb);
    v8h hi = *(const v8h*)(rowp + kb + 8);
    v16h a;
#pragma unroll
    for (int i = 0; i < 8; ++i) { a[i] = lo[i]; a[i + 8] = hi[i]; }
    return a;
}

// B fragment from LDS, undoing the XOR bank swizzle.
__device__ __forceinline__ v16h load_bfrag(const _Float16* fragp, int lane) {
    int u0 = 2 * lane;
    int u1 = 2 * lane + 1;
    int p0 = u0 ^ (u0 >> 4);
    int p1 = u1 ^ (u1 >> 4);
    v8h lo = *(const v8h*)(fragp + p0 * 8);
    v8h hi = *(const v8h*)(fragp + p1 * 8);
    v16h b;
#pragma unroll
    for (int i = 0; i < 8; ++i) { b[i] = lo[i]; b[i + 8] = hi[i]; }
    return b;
}

__global__ __launch_bounds__(256)
void deform_kernel(const float* __restrict__ pts,
                   const float* __restrict__ time_emb,
                   const float* __restrict__ shs_emb,
                   const float* __restrict__ space_planes,
                   const float* __restrict__ time_planes,
                   const _Float16* __restrict__ wsW,
                   const float* __restrict__ b_enc,
                   const float* __restrict__ b1,
                   const float* __restrict__ b2,
                   float* __restrict__ out_shs, int N, int numTiles) {
    extern __shared__ __align__(32) char smem[];
    _Float16* sW = (_Float16*)(smem + SW_LDS_OFF);   // 172032 B weight frags
    _Float16* sX = (_Float16*)(smem + SX_LDS_OFF);   // 256 x 36 halfs
    _Float16* sH = (_Float16*)(smem + SH_LDS_OFF);   // 8 waves x 32 x 264 halfs

    const int tid  = threadIdx.x;
    const int lane = tid & 31;
    const int w    = tid >> 5;                       // wave 0..7

    // ---- Stage all MLP weights into LDS once per block via TDM ----
    if (w == 0) {
        unsigned      ldsAddr = (unsigned)(size_t)(void*)sW;   // LDS aperture: low 32 bits
        unsigned long long ga = (unsigned long long)(size_t)wsW;
        u32x4 g0;
        g0[0] = 1u;                                   // count=1, user descriptor
        g0[1] = ldsAddr;                              // lds_addr
        g0[2] = (unsigned)(ga & 0xFFFFFFFFu);         // global_addr[31:0]
        g0[3] = (unsigned)((ga >> 32) & 0x01FFFFFFu)  // global_addr[56:32]
              | 0x80000000u;                          // type=2 ("image")
        u32x8 g1;
        g1[0] = 3u << 16;                             // data_size=3 (8B elements)
        g1[1] = (unsigned)(WS_E8 & 0xFFFF) << 16;     // tensor_dim0[15:0]
        g1[2] = 1u << 16;                             // tensor_dim0[31:16]=0, tensor_dim1=1
        g1[3] = (unsigned)(WS_E8 & 0xFFFF) << 16;     // tile_dim0 = 21504
        g1[4] = 1u;                                   // tile_dim1 = 1
        g1[5] = (unsigned)WS_E8;                      // tensor_dim0_stride[31:0]
        g1[6] = 0u;
        g1[7] = 0u;
        asm volatile("tensor_load_to_lds %0, %1" :: "s"(g0), "s"(g1) : "memory");
        __builtin_amdgcn_s_wait_tensorcnt(0);
    }
    __syncthreads();

    const _Float16* wencS = sW + WENC_OFF;
    const _Float16* w1S   = sW + W1_OFF;
    const _Float16* w2S   = sW + W2_OFF;
    _Float16* sHw = sH + w * 32 * SH_STRIDE;         // this wave's 32-row H tile
    const int lrow = lane & 15;
    const int lhi  = lane >> 4;                      // 0 or 1
    const int kb   = lhi * 16;
    const v8f zero8 = { 0.f, 0.f, 0.f, 0.f, 0.f, 0.f, 0.f, 0.f };

    for (int tile = blockIdx.x; tile < numTiles; tile += gridDim.x) {
        const int base = tile * 256;

        // ---------------- Phase 1: plane features -> sX (f16), 1 row/thread ----------------
        {
            int row = base + tid; if (row > N - 1) row = N - 1;
            float px = pts[row * 3 + 0];
            float py = pts[row * 3 + 1];
            float pz = pts[row * 3 + 2];
            float xn = clampf(px * (1.0f / 1.6f), -1.f, 1.f) * 0.5f + 0.5f;
            float yn = clampf(py * (1.0f / 1.6f), -1.f, 1.f) * 0.5f + 0.5f;
            float zn = clampf(pz * (1.0f / 1.6f), -1.f, 1.f) * 0.5f + 0.5f;
            float t  = clampf(time_emb[row], 0.f, 1.f);

            const float us[6] = { xn, xn, yn, xn, yn, zn };
            const float vs[6] = { yn, zn, zn, t,  t,  t  };
            const int PS = PLANE_R * PLANE_R * PLANE_F;
            const float* pl[6] = { space_planes, space_planes + PS, space_planes + 2 * PS,
                                   time_planes,  time_planes  + PS, time_planes  + 2 * PS };
            const float* c00[6];
            float w00[6], w01[6], w10[6], w11[6];
#pragma unroll
            for (int i = 0; i < 6; ++i) {
                float x = us[i] * (float)(PLANE_R - 1);
                float y = vs[i] * (float)(PLANE_R - 1);
                float x0f = clampf(floorf(x), 0.f, (float)(PLANE_R - 2));
                float y0f = clampf(floorf(y), 0.f, (float)(PLANE_R - 2));
                int x0 = (int)x0f, y0 = (int)y0f;
                float fx = x - x0f, fy = y - y0f;
                c00[i] = pl[i] + ((size_t)y0 * PLANE_R + x0) * PLANE_F;
                w00[i] = (1.f - fx) * (1.f - fy);
                w01[i] = fx * (1.f - fy);
                w10[i] = (1.f - fx) * fy;
                w11[i] = fx * fy;
            }
            for (int f = 0; f < 32; ++f) {
                float prod = 1.0f;
#pragma unroll
                for (int i = 0; i < 6; ++i) {
                    const float* b = c00[i] + f;
                    float v = b[0] * w00[i] + b[PLANE_F] * w01[i] +
                              b[PLANE_F * PLANE_R] * w10[i] +
                              b[PLANE_F * PLANE_R + PLANE_F] * w11[i];
                    prod *= v;
                }
                sX[tid * SX_STRIDE + f] = (_Float16)prod;
            }
        }
        __syncthreads();

        // ---------------- Layer 1: X(32x32) @ w_enc(32x256), 2 M-tiles/wave ----------------
        {
            v16h a0 = load_afrag(&sX[(w * 32 + lrow) * SX_STRIDE], kb);
            v16h a1 = load_afrag(&sX[(w * 32 + 16 + lrow) * SX_STRIDE], kb);
#pragma unroll
            for (int nt = 0; nt < 16; ++nt) {
                v16h b = load_bfrag(wencS + nt * 512, lane);
                v8f c0 = zero8, c1 = zero8;
                c0 = __builtin_amdgcn_wmma_f32_16x16x32_f16(false, a0, false, b,
                                                            (short)0, c0, false, false);
                c1 = __builtin_amdgcn_wmma_f32_16x16x32_f16(false, a1, false, b,
                                                            (short)0, c1, false, false);
                int col    = nt * 16 + lrow;
                float bias = b_enc[col];
#pragma unroll
                for (int r = 0; r < 8; ++r) {
                    int M = r + (lhi << 3);
                    float h0 = fmaxf(c0[r] + bias, 0.f);
                    float h1 = fmaxf(c1[r] + bias, 0.f);
                    sHw[M * SH_STRIDE + col]        = (_Float16)h0;
                    sHw[(16 + M) * SH_STRIDE + col] = (_Float16)h1;
                }
            }
        }
        __syncthreads();

        // ---------------- Layer 2: H(32x256) @ w1(256x256), acc in regs ----------------
        v8f acc2[2][16];
#pragma unroll
        for (int nt = 0; nt < 16; ++nt) { acc2[0][nt] = zero8; acc2[1][nt] = zero8; }

        for (int kt = 0; kt < 8; ++kt) {
            v16h a0 = load_afrag(&sHw[lrow * SH_STRIDE], kt * 32 + kb);
            v16h a1 = load_afrag(&sHw[(16 + lrow) * SH_STRIDE], kt * 32 + kb);
#pragma unroll
            for (int nt = 0; nt < 16; ++nt) {
                v16h b = load_bfrag(w1S + (kt * 16 + nt) * 512, lane);
                acc2[0][nt] = __builtin_amdgcn_wmma_f32_16x16x32_f16(false, a0, false, b,
                                                                     (short)0, acc2[0][nt],
                                                                     false, false);
                acc2[1][nt] = __builtin_amdgcn_wmma_f32_16x16x32_f16(false, a1, false, b,
                                                                     (short)0, acc2[1][nt],
                                                                     false, false);
            }
        }
        __syncthreads();

        // bias + relu, overwrite sHw with H2
#pragma unroll
        for (int nt = 0; nt < 16; ++nt) {
            int col    = nt * 16 + lrow;
            float bias = b1[col];
#pragma unroll
            for (int r = 0; r < 8; ++r) {
                int M = r + (lhi << 3);
                float h0 = fmaxf(acc2[0][nt][r] + bias, 0.f);
                float h1 = fmaxf(acc2[1][nt][r] + bias, 0.f);
                sHw[M * SH_STRIDE + col]        = (_Float16)h0;
                sHw[(16 + M) * SH_STRIDE + col] = (_Float16)h1;
            }
        }
        __syncthreads();

        // ---------------- Layer 3: H2(32x256) @ w2(256x48) ----------------
        v8f acc3[2][3];
#pragma unroll
        for (int nt = 0; nt < 3; ++nt) { acc3[0][nt] = zero8; acc3[1][nt] = zero8; }

        for (int kt = 0; kt < 8; ++kt) {
            v16h a0 = load_afrag(&sHw[lrow * SH_STRIDE], kt * 32 + kb);
            v16h a1 = load_afrag(&sHw[(16 + lrow) * SH_STRIDE], kt * 32 + kb);
#pragma unroll
            for (int nt = 0; nt < 3; ++nt) {
                v16h b = load_bfrag(w2S + (kt * 3 + nt) * 512, lane);
                acc3[0][nt] = __builtin_amdgcn_wmma_f32_16x16x32_f16(false, a0, false, b,
                                                                     (short)0, acc3[0][nt],
                                                                     false, false);
                acc3[1][nt] = __builtin_amdgcn_wmma_f32_16x16x32_f16(false, a1, false, b,
                                                                     (short)0, acc3[1][nt],
                                                                     false, false);
            }
        }

        // epilogue: shs = shs_emb + d
#pragma unroll
        for (int nt = 0; nt < 3; ++nt) {
            int col    = nt * 16 + lrow;
            float bias = b2[col];
#pragma unroll
            for (int mt = 0; mt < 2; ++mt) {
#pragma unroll
                for (int r = 0; r < 8; ++r) {
                    int M  = r + (lhi << 3);
                    int rg = base + w * 32 + mt * 16 + M;
                    if (rg < N) {
                        size_t o = (size_t)rg * 48 + col;
                        out_shs[o] = shs_emb[o] + acc3[mt][nt][r] + bias;
                    }
                }
            }
        }
        __syncthreads();
    }
}

extern "C" void kernel_launch(void* const* d_in, const int* in_sizes, int n_in,
                              void* d_out, int out_size, void* d_ws, size_t ws_size,
                              hipStream_t stream) {
    const float* pts     = (const float*)d_in[0];
    const float* rot     = (const float*)d_in[1];
    const float* shs     = (const float*)d_in[3];
    const float* time_e  = (const float*)d_in[5];
    const float* h_emb   = (const float*)d_in[6];
    const float* sp      = (const float*)d_in[7];
    const float* tp      = (const float*)d_in[8];
    const float* w_enc   = (const float*)d_in[9];
    const float* b_enc   = (const float*)d_in[10];
    const float* w1      = (const float*)d_in[11];
    const float* b1      = (const float*)d_in[12];
    const float* w2      = (const float*)d_in[13];
    const float* b2      = (const float*)d_in[14];

    const int N = in_sizes[0] / 3;
    float* out = (float*)d_out;
    _Float16* wsH = (_Float16*)d_ws;

    // 1) f16 convert + WMMA-B fragment swizzle into d_ws
    wconvert_kernel<<<(WS_HALFS + 255) / 256, 256, 0, stream>>>(w_enc, w1, w2, wsH);

    // 2) pass-through outputs
    passthrough_kernel<<<(N + 255) / 256, 256, 0, stream>>>(pts, rot, h_emb, out, N);

    // 3) fused plane-sample + 3-layer WMMA MLP, persistent blocks, TDM weight staging,
    //    M=32 register blocking per wave (halves B-fragment LDS traffic)
    const int numTiles = (N + 255) / 256;
    int blocks = numTiles < 1024 ? numTiles : 1024;
    deform_kernel<<<blocks, 256, SMEM_BYTES, stream>>>(
        pts, time_e, shs, sp, tp, wsH, b_enc, b1, b2,
        out + (size_t)8 * N, N, numTiles);
}
